// VmfVQ_11897059410180
// MI455X (gfx1250) — matrix operand order repaired and checked by reading, not compiled, not run on hardware
//
#include <hip/hip_runtime.h>
#include <hip/hip_bf16.h>
#include <math.h>

typedef __attribute__((ext_vector_type(16))) _Float16 v16h;
typedef __attribute__((ext_vector_type(8)))  float    v8f;

#define N_ROWS 65536
#define V_DIM  2048
#define E_DIM  1024
#define KAPPA_Q 1.05f      /* exp(-2.995732273553991)+1 = 0.05+1 */
#define INV_T   2.0f       /* 1/0.5 */
#define GEPS    1e-10f

#define WMMA_F16(a, b, c) \
  __builtin_amdgcn_wmma_f32_16x16x32_f16(false, (a), false, (b), (short)0, (c), false, false)

union HalfVec { uint4 u[2]; v16h v; };

// A-matrix 16x32 f16 fragment (ISA 7.12.2): lane m (&15) holds row m;
// lo lanes: K {0..7,16..23}, hi lanes: K {8..15,24..31} of this k-step.
__device__ inline v16h load_fragA(const _Float16* row, int ks, int hi) {
  const _Float16* p = row + ks * 32 + (hi ? 8 : 0);
  HalfVec hv;
  hv.u[0] = *(const uint4*)(p);        // K kb..kb+7
  hv.u[1] = *(const uint4*)(p + 16);   // K kb+16..kb+23
  return hv.v;
}

// B-matrix 32x16 f16 fragment: lane n (&15) holds column n;
// lo lanes: K {0..7,16..23}, hi lanes: K {8..15,24..31} of this k-step.
__device__ inline v16h load_fragB(const _Float16* col, int ks, int hi) {
  const _Float16* p = col + ks * 32 + (hi ? 8 : 0);
  HalfVec hv;
  hv.u[0] = *(const uint4*)(p);
  hv.u[1] = *(const uint4*)(p + 16);
  return hv.v;
}

// ---- Kernel 1: normalize emb rows -> embn f16 [V,E] and embnT f16 [E,V] ----
__global__ void __launch_bounds__(256)
emb_prep_kernel(const float* __restrict__ emb,
                _Float16* __restrict__ embn,
                _Float16* __restrict__ embnT) {
  __shared__ float red[256];
  const int v = blockIdx.x;
  const int t = threadIdx.x;
  const float4 f4 = *(const float4*)(emb + (size_t)v * E_DIM + t * 4);
  red[t] = f4.x * f4.x + f4.y * f4.y + f4.z * f4.z + f4.w * f4.w;
  __syncthreads();
  for (int s = 128; s > 0; s >>= 1) {
    if (t < s) red[t] += red[t + s];
    __syncthreads();
  }
  const float rn = 1.0f / fmaxf(sqrtf(red[0]), 1e-12f);
  union { _Float16 h[4]; uint2 u; } o;
  o.h[0] = (_Float16)(f4.x * rn); o.h[1] = (_Float16)(f4.y * rn);
  o.h[2] = (_Float16)(f4.z * rn); o.h[3] = (_Float16)(f4.w * rn);
  *(uint2*)(embn + (size_t)v * E_DIM + t * 4) = o.u;
  #pragma unroll
  for (int i = 0; i < 4; ++i)
    embnT[(size_t)(t * 4 + i) * V_DIM + v] = o.h[i];
}

// ---- Kernel 2: fused  logits -> gumbel-softmax -> z_q  (16 rows / WG) ----
__global__ void __launch_bounds__(256)
vq_fused_kernel(const float* __restrict__ z,
                const float* __restrict__ U,
                const _Float16* __restrict__ embn,
                const _Float16* __restrict__ embnT,
                float* __restrict__ tokens_out,
                float* __restrict__ zq_out) {
  __shared__ _Float16 zn_s[16 * E_DIM];   // 32 KB: normalized z tile (f16)
  __shared__ _Float16 sbuf[16 * V_DIM];   // 64 KB: z f32 staging -> s f16 -> tokens f16
  __shared__ float red[16][17];

  const int tid  = threadIdx.x;
  const int row0 = blockIdx.x * 16;
  const int lane = tid & 31;
  const int wave = tid >> 5;
  const int ln   = lane & 15;
  const int hi   = lane >> 4;

  // ---------- Phase 0a: async bulk copy of the contiguous 64KB z tile -> LDS ----------
  {
    float* zstage = (float*)sbuf;                       // 16*1024 f32
    const uint32_t lds_base = (uint32_t)(uintptr_t)zstage;
    const uint64_t src_base = (uint64_t)(uintptr_t)(z + (size_t)row0 * E_DIM);
    #pragma unroll
    for (int j = 0; j < 16; ++j) {
      const uint32_t chunk = (uint32_t)(tid + 256 * j); // 4096 x 16B = 64KB
      const uint32_t dst   = lds_base + chunk * 16u;
      const uint64_t srca  = src_base + (uint64_t)chunk * 16u;
      asm volatile("global_load_async_to_lds_b128 %0, %1, off"
                   :: "v"(dst), "v"(srca) : "memory");
    }
    asm volatile("s_wait_asynccnt 0x0" ::: "memory");
  }
  __syncthreads();

  // ---------- Phase 0b: L2-normalize 16 rows in LDS, emit f16 zn ----------
  {
    float* zstage = (float*)sbuf;
    const int r  = tid >> 4;
    const int c0 = tid & 15;
    float ss = 0.f;
    #pragma unroll
    for (int j = 0; j < 16; ++j) {
      const int idx = c0 + 16 * j;
      float4 f4 = *(const float4*)(zstage + r * E_DIM + idx * 4);
      ss += f4.x * f4.x + f4.y * f4.y + f4.z * f4.z + f4.w * f4.w;
    }
    red[r][c0] = ss;
    __syncthreads();
    if (c0 == 0) {
      float s = 0.f;
      #pragma unroll
      for (int i = 0; i < 16; ++i) s += red[r][i];
      red[r][16] = 1.0f / fmaxf(sqrtf(s), 1e-12f);
    }
    __syncthreads();
    const float rn = red[r][16];
    #pragma unroll
    for (int j = 0; j < 16; ++j) {
      const int idx = c0 + 16 * j;
      float4 f4 = *(const float4*)(zstage + r * E_DIM + idx * 4);
      union { _Float16 h[4]; uint2 u; } o;
      o.h[0] = (_Float16)(f4.x * rn); o.h[1] = (_Float16)(f4.y * rn);
      o.h[2] = (_Float16)(f4.z * rn); o.h[3] = (_Float16)(f4.w * rn);
      *(uint2*)(zn_s + r * E_DIM + idx * 4) = o.u;
    }
  }
  __syncthreads();

  // ---------- Phase 1: s = (kappa * zn@embn^T + g)/T  -> sbuf (f16) ----------
  // Wave owns 256 logit columns; 4 column tiles per K-pass so one A fragment
  // feeds 4 WMMAs (4x less LDS traffic per matrix op).
  {
    const int colbase = wave * 256;
    const _Float16* arow = zn_s + ln * E_DIM;
    for (int cg = 0; cg < 4; ++cg) {
      const int col0 = colbase + cg * 64;
      const _Float16* bbase = embn + (size_t)(col0 + ln) * E_DIM;
      v8f acc[4] = {};
      #pragma unroll 2
      for (int ks = 0; ks < 32; ++ks) {
        v16h a = load_fragA(arow, ks, hi);
        #pragma unroll
        for (int t = 0; t < 4; ++t) {
          v16h b = load_fragB(bbase + (size_t)(t * 16) * E_DIM, ks, hi);
          acc[t] = WMMA_F16(a, b, acc[t]);
        }
      }
      #pragma unroll
      for (int t = 0; t < 4; ++t) {
        const int cc = col0 + t * 16 + ln;
        #pragma unroll
        for (int i = 0; i < 8; ++i) {
          const int m = i + 8 * hi;               // D layout: M = vgpr + 8*(lane>=16)
          const float u = U[(size_t)(row0 + m) * V_DIM + cc];
          const float g = -__logf(-__logf(u + GEPS) + GEPS);
          sbuf[m * V_DIM + cc] = (_Float16)((KAPPA_Q * acc[t][i] + g) * INV_T);
        }
      }
    }
  }
  __syncthreads();

  // ---------- Phase 2: softmax over V; tokens -> global f32 + LDS f16 ----------
  {
    const int r  = tid >> 4;
    const int c0 = tid & 15;
    float mx = -1e30f;
    for (int j = 0; j < 128; ++j)
      mx = fmaxf(mx, (float)sbuf[r * V_DIM + c0 + 16 * j]);
    red[r][c0] = mx;
    __syncthreads();
    if (c0 == 0) {
      float m2 = red[r][0];
      #pragma unroll
      for (int i = 1; i < 16; ++i) m2 = fmaxf(m2, red[r][i]);
      red[r][16] = m2;
    }
    __syncthreads();
    mx = red[r][16];
    float sum = 0.f;
    for (int j = 0; j < 128; ++j)
      sum += __expf((float)sbuf[r * V_DIM + c0 + 16 * j] - mx);
    __syncthreads();                 // everyone has read red[r][16] before reuse
    red[r][c0] = sum;
    __syncthreads();
    if (c0 == 0) {
      float s2 = 0.f;
      #pragma unroll
      for (int i = 0; i < 16; ++i) s2 += red[r][i];
      red[r][16] = 1.0f / s2;
    }
    __syncthreads();
    const float inv = red[r][16];
    for (int j = 0; j < 128; ++j) {
      const int c = c0 + 16 * j;
      const float p = __expf((float)sbuf[r * V_DIM + c] - mx) * inv;
      tokens_out[(size_t)(row0 + r) * V_DIM + c] = p;
      sbuf[r * V_DIM + c] = (_Float16)p;       // in-place: tokens f16 for GEMM2
    }
  }
  __syncthreads();

  // ---------- Phase 3: z_q = tokens @ embn ;  K = V = 2048 (64 wmma steps) ----------
  // Wave owns 128 z_q columns; again 4 tiles per K-pass.
  {
    const _Float16* trow = sbuf + ln * V_DIM;
    for (int cg = 0; cg < 2; ++cg) {
      const int ecol0 = wave * 128 + cg * 64;
      const _Float16* bbase = embnT + (size_t)(ecol0 + ln) * V_DIM;
      v8f acc[4] = {};
      #pragma unroll 2
      for (int ks = 0; ks < 64; ++ks) {
        v16h a = load_fragA(trow, ks, hi);
        #pragma unroll
        for (int t = 0; t < 4; ++t) {
          v16h b = load_fragB(bbase + (size_t)(t * 16) * V_DIM, ks, hi);
          acc[t] = WMMA_F16(a, b, acc[t]);
        }
      }
      #pragma unroll
      for (int t = 0; t < 4; ++t) {
        const int ec = ecol0 + t * 16 + ln;
        #pragma unroll
        for (int i = 0; i < 8; ++i) {
          const int m = i + 8 * hi;
          zq_out[(size_t)(row0 + m) * E_DIM + ec] = acc[t][i];
        }
      }
    }
  }
}

extern "C" void kernel_launch(void* const* d_in, const int* in_sizes, int n_in,
                              void* d_out, int out_size, void* d_ws, size_t ws_size,
                              hipStream_t stream) {
  const float* z     = (const float*)d_in[0];
  const float* emb_w = (const float*)d_in[1];
  const float* U     = (const float*)d_in[2];
  float* tokens_out  = (float*)d_out;                               // [N,V]
  float* zq_out      = (float*)d_out + (size_t)N_ROWS * V_DIM;      // [N,E]
  _Float16* embn     = (_Float16*)d_ws;                             // [V,E] f16, 4 MB
  _Float16* embnT    = (_Float16*)d_ws + (size_t)V_DIM * E_DIM;     // [E,V] f16, 4 MB

  emb_prep_kernel<<<V_DIM, 256, 0, stream>>>(emb_w, embn, embnT);
  vq_fused_kernel<<<N_ROWS / 16, 256, 0, stream>>>(z, U, embn, embnT,
                                                   tokens_out, zq_out);
}
